// MixtureOfHMM_54425825575669
// MI455X (gfx1250) — compile-verified, matrix operand order checked by reading, not run on hardware
//
#include <hip/hip_runtime.h>
#include <hip/hip_bf16.h>
#include <math.h>

typedef __attribute__((ext_vector_type(16))) _Float16 v16h;
typedef __attribute__((ext_vector_type(8)))  float    v8f;

#define B_ 32
#define T_ 512
#define G_ 32000
#define E_ 256
#define M_ 16
#define S_ 128

// ---------------------------------------------------------------------------
// Kernel 1: mean_emb[b,e] = mean_t embed_W[x[b,t], e]
// 32 blocks x 256 threads; each t-iteration reads one contiguous 1KB row.
// ---------------------------------------------------------------------------
__global__ void mean_emb_kernel(const int* __restrict__ x,
                                const float* __restrict__ embed_W,
                                float* __restrict__ mean_out) {
    int b = blockIdx.x;
    int e = threadIdx.x;
    const int* xb = x + b * T_;
    float acc = 0.f;
    for (int t = 0; t < T_; ++t) {
        int g = xb[t];
        acc += embed_W[(size_t)g * E_ + e];
    }
    mean_out[b * E_ + e] = acc * (1.0f / T_);
}

// ---------------------------------------------------------------------------
// Kernel 2: logits[b,g] = mean_emb[b,:] . vocab_W[g,:] + vocab_b[g]
// via v_wmma_f32_16x16x32_f16. One wave per 16-column tile of G, both
// 16-row tiles of B=32, K=256 in 8 steps.
// A 16x32 f16 layout: lane&15 = M, VGPR pair j -> K = 2j + (j>=4?8:0) + 8*(lane>>4)
// B 32x16 f16 layout: mirrored with lane&15 = N.
// D f32: M = r + 8*(lane>>4), N = lane&15.
// ---------------------------------------------------------------------------
__global__ void __launch_bounds__(256) logits_wmma_kernel(
        const float* __restrict__ mean_emb,
        const float* __restrict__ vocab_W,
        const float* __restrict__ vocab_b,
        float* __restrict__ logits) {
    int wave = threadIdx.x >> 5;
    int lane = threadIdx.x & 31;
    int tile = blockIdx.x * 8 + wave;      // 0..1999
    int g0   = tile * 16;
    int nm   = lane & 15;
    int half = lane >> 4;

    const float* arow0 = mean_emb + (size_t)nm * E_;          // rows 0..15
    const float* arow1 = mean_emb + (size_t)(16 + nm) * E_;   // rows 16..31
    const float* brow  = vocab_W + (size_t)(g0 + nm) * E_;    // col g0+nm

    v8f acc0 = {};
    v8f acc1 = {};
    for (int s = 0; s < 8; ++s) {
        int kb = s * 32;
        v16h a0, a1, bb;
#pragma unroll
        for (int j = 0; j < 8; ++j) {
            int K = kb + 2 * j + ((j & 4) ? 8 : 0) + 8 * half; // even pair base
            float2 fa0 = *(const float2*)(arow0 + K);
            float2 fa1 = *(const float2*)(arow1 + K);
            float2 fb  = *(const float2*)(brow + K);
            a0[2 * j]     = (_Float16)fa0.x;
            a0[2 * j + 1] = (_Float16)fa0.y;
            a1[2 * j]     = (_Float16)fa1.x;
            a1[2 * j + 1] = (_Float16)fa1.y;
            bb[2 * j]     = (_Float16)fb.x;
            bb[2 * j + 1] = (_Float16)fb.y;
        }
        if (s < 7) __builtin_prefetch(brow + kb + 32, 0, 0);   // global_prefetch_b8
        acc0 = __builtin_amdgcn_wmma_f32_16x16x32_f16(false, a0, false, bb,
                                                      (short)0, acc0, false, false);
        acc1 = __builtin_amdgcn_wmma_f32_16x16x32_f16(false, a1, false, bb,
                                                      (short)0, acc1, false, false);
    }
    float bias = vocab_b[g0 + nm];
#pragma unroll
    for (int r = 0; r < 8; ++r) {
        int m = r + 8 * half;
        logits[(size_t)m * G_ + g0 + nm]        = acc0[r] + bias;
        logits[(size_t)(16 + m) * G_ + g0 + nm] = acc1[r] + bias;
    }
}

// ---------------------------------------------------------------------------
// Kernel 3: lseb[b] = logsumexp_g logits[b,:]
// ---------------------------------------------------------------------------
__global__ void lse_kernel(const float* __restrict__ logits,
                           float* __restrict__ lseb) {
    __shared__ float red[256];
    int b = blockIdx.x, tid = threadIdx.x;
    const float* row = logits + (size_t)b * G_;
    float mx = -INFINITY;
    for (int g = tid; g < G_; g += 256) mx = fmaxf(mx, row[g]);
    red[tid] = mx;
    __syncthreads();
    for (int s = 128; s > 0; s >>= 1) {
        if (tid < s) red[tid] = fmaxf(red[tid], red[tid + s]);
        __syncthreads();
    }
    mx = red[0];
    __syncthreads();
    float sum = 0.f;
    for (int g = tid; g < G_; g += 256) sum += __expf(row[g] - mx);
    red[tid] = sum;
    __syncthreads();
    for (int s = 128; s > 0; s >>= 1) {
        if (tid < s) red[tid] += red[tid + s];
        __syncthreads();
    }
    if (tid == 0) lseb[b] = mx + __logf(red[0]);
}

// ---------------------------------------------------------------------------
// Kernel 4: esum[b] = sum_t logits[b, x[b,t]] - T * lseb[b]
// ---------------------------------------------------------------------------
__global__ void esum_kernel(const float* __restrict__ logits,
                            const int* __restrict__ x,
                            const float* __restrict__ lseb,
                            float* __restrict__ esum) {
    __shared__ float red[256];
    int b = blockIdx.x, tid = threadIdx.x;
    const int* xb = x + b * T_;
    const float* row = logits + (size_t)b * G_;
    float acc = 0.f;
    for (int t = tid; t < T_; t += 256) acc += row[xb[t]];
    red[tid] = acc;
    __syncthreads();
    for (int s = 128; s > 0; s >>= 1) {
        if (tid < s) red[tid] += red[tid + s];
        __syncthreads();
    }
    if (tid == 0) esum[b] = red[0] - (float)T_ * lseb[b];
}

// ---------------------------------------------------------------------------
// Kernel 5: logTt[m][s_to][s_from] = log_softmax over s_from of 100*trans
// (transposed store so the chain reads contiguous s_from rows).
// 16 blocks x 128 threads; each thread owns one s_to column in LDS.
// ---------------------------------------------------------------------------
__global__ void logT_kernel(const float* __restrict__ transition,
                            float* __restrict__ logTt) {
    __shared__ float col[S_ * S_];   // col[sf*S_ + st], 64 KB
    int m = blockIdx.x, st = threadIdx.x;
    const float* tm = transition + (size_t)m * S_ * S_;
    for (int sf = 0; sf < S_; ++sf)
        col[sf * S_ + st] = 100.0f * tm[sf * S_ + st];   // coalesced over st
    float mx = -INFINITY;
    for (int sf = 0; sf < S_; ++sf) mx = fmaxf(mx, col[sf * S_ + st]);
    float sum = 0.f;
    for (int sf = 0; sf < S_; ++sf) sum += __expf(col[sf * S_ + st] - mx);
    float lse = mx + __logf(sum);
    float* out = logTt + ((size_t)m * S_ + st) * S_;
    for (int sf = 0; sf < S_; ++sf) out[sf] = col[sf * S_ + st] - lse;
}

// ---------------------------------------------------------------------------
// Kernel 6: 512-step lse-matrix chain per mixture component m.
// 16 blocks x 1024 threads; W resident in LDS (stride 132 avoids bank
// conflicts); 8 lanes per output state combined via width-8 shfl_xor
// (max,sum) lse-merge. Outputs Pm[m] = LSE_s(C_T[m,s]/T).
// ---------------------------------------------------------------------------
#define WSTRIDE 132
__global__ void __launch_bounds__(1024) chain_kernel(
        const float* __restrict__ logTt,
        const float* __restrict__ init_dist,
        float* __restrict__ Pm) {
    __shared__ float W[S_ * WSTRIDE];   // W[st*WSTRIDE + sf]
    __shared__ float C[S_];
    __shared__ float Cn[S_];
    __shared__ float red[S_];
    int m = blockIdx.x, tid = threadIdx.x;

    const float* src = logTt + (size_t)m * S_ * S_;
    for (int i = tid; i < S_ * S_; i += 1024) {
        int st = i >> 7, sf = i & 127;
        W[st * WSTRIDE + sf] = src[i];
    }
    if (tid < S_) red[tid] = 100.0f * init_dist[m * S_ + tid];
    __syncthreads();
    if (tid == 0) {   // cheap one-time serial lse of 128 values
        float mx = -INFINITY;
        for (int s = 0; s < S_; ++s) mx = fmaxf(mx, red[s]);
        float sum = 0.f;
        for (int s = 0; s < S_; ++s) sum += __expf(red[s] - mx);
        Cn[0] = mx + __logf(sum);
    }
    __syncthreads();
    if (tid < S_) C[tid] = red[tid] - Cn[0];
    __syncthreads();

    int col = tid >> 3;    // s_to
    int sub = tid & 7;
    const float* wrow = &W[col * WSTRIDE + sub * 16];
    const float* cp   = &C[sub * 16];
    for (int t = 0; t < T_; ++t) {
        float mx = -INFINITY;
#pragma unroll
        for (int i = 0; i < 16; ++i) mx = fmaxf(mx, wrow[i] + cp[i]);
        float sum = 0.f;
#pragma unroll
        for (int i = 0; i < 16; ++i) sum += __expf(wrow[i] + cp[i] - mx);
#pragma unroll
        for (int off = 1; off < 8; off <<= 1) {
            float mo = __shfl_xor(mx, off, 8);
            float so = __shfl_xor(sum, off, 8);
            float nm = fmaxf(mx, mo);
            sum = sum * __expf(mx - nm) + so * __expf(mo - nm);
            mx = nm;
        }
        if (sub == 0) Cn[col] = mx + __logf(sum);
        __syncthreads();
        if (tid < S_) C[tid] = Cn[tid];
        __syncthreads();
    }
    if (tid == 0) {
        float mx = -INFINITY;
        for (int s = 0; s < S_; ++s) mx = fmaxf(mx, C[s] * (1.0f / T_));
        float sum = 0.f;
        for (int s = 0; s < S_; ++s) sum += __expf(C[s] * (1.0f / T_) - mx);
        Pm[m] = mx + __logf(sum);
    }
}

// ---------------------------------------------------------------------------
// Kernel 7: out[b] = LSE_m(Pm[m]) + esum[b]/T   (single wave)
// ---------------------------------------------------------------------------
__global__ void final_kernel(const float* __restrict__ Pm,
                             const float* __restrict__ esum,
                             float* __restrict__ out) {
    int tid = threadIdx.x;   // 0..31
    float v = (tid < M_) ? Pm[tid] : -INFINITY;
    float mx = v;
    for (int off = 16; off > 0; off >>= 1) mx = fmaxf(mx, __shfl_xor(mx, off, 32));
    float s = (tid < M_) ? __expf(v - mx) : 0.f;
    for (int off = 16; off > 0; off >>= 1) s += __shfl_xor(s, off, 32);
    float K = mx + __logf(s);
    out[tid] = K + esum[tid] * (1.0f / T_);
}

// ---------------------------------------------------------------------------
// Inputs (setup_inputs order):
// 0 zi (int[B], unused), 1 x (int[B*T]), 2 embed_W (f32[G*E]),
// 3 vocab_W (f32[G*E]), 4 vocab_b (f32[G]), 5 init_dist (f32[M*S]),
// 6 transition (f32[M*S*S]), 7 state_vect (f32[M*S*E], unused)
// ---------------------------------------------------------------------------
extern "C" void kernel_launch(void* const* d_in, const int* in_sizes, int n_in,
                              void* d_out, int out_size, void* d_ws, size_t ws_size,
                              hipStream_t stream) {
    const int*   x          = (const int*)d_in[1];
    const float* embed_W    = (const float*)d_in[2];
    const float* vocab_W    = (const float*)d_in[3];
    const float* vocab_b    = (const float*)d_in[4];
    const float* init_dist  = (const float*)d_in[5];
    const float* transition = (const float*)d_in[6];

    float* ws     = (float*)d_ws;
    float* mean   = ws;                                 // 32*256
    float* logits = mean + (size_t)B_ * E_;             // 32*32000
    float* lseb   = logits + (size_t)B_ * G_;           // 32
    float* esum   = lseb + B_;                          // 32
    float* logTt  = esum + B_;                          // 16*128*128
    float* Pm     = logTt + (size_t)M_ * S_ * S_;       // 16
    // total ~5.2 MB of ws

    mean_emb_kernel<<<B_, E_, 0, stream>>>(x, embed_W, mean);
    logits_wmma_kernel<<<G_ / 16 / 8, 256, 0, stream>>>(mean, vocab_W, vocab_b, logits);
    lse_kernel<<<B_, 256, 0, stream>>>(logits, lseb);
    esum_kernel<<<B_, 256, 0, stream>>>(logits, x, lseb, esum);
    logT_kernel<<<M_, S_, 0, stream>>>(transition, logTt);
    chain_kernel<<<M_, 1024, 0, stream>>>(logTt, init_dist, Pm);
    final_kernel<<<1, 32, 0, stream>>>(Pm, esum, (float*)d_out);
}
